// regDGCNN_seg_584115552470
// MI455X (gfx1250) — compile-verified, matrix operand order checked by reading, not run on hardware
//
#include <hip/hip_runtime.h>

typedef __attribute__((ext_vector_type(16))) _Float16 v16h;
typedef __attribute__((ext_vector_type(8)))  _Float16 v8h;
typedef __attribute__((ext_vector_type(8)))  float    v8f;

#define NPTS 8192
#define KNN  20
#define NEGS 0.2f

// ---------------------------------------------------------------------------
// WMMA helpers (CDNA5 wave32, 16x16x32 f16 -> f32)
// A 16x32: lane L%16 = row M; lane>>4 selects K-halves {0..7,16..23} vs {8..15,24..31}
// B 32x16: lane L%16 = col N; same K-slice rule.
// C 16x16 f32: lane L%16 = col N; VGPR r = row r + 8*(L>>4).
// ---------------------------------------------------------------------------
__device__ __forceinline__ v8f wmma32(v16h a, v16h b, v8f c) {
  return __builtin_amdgcn_wmma_f32_16x16x32_f16(false, a, false, b, (short)0, c, false, false);
}

// p = row/col channel base + kc*32 ; hl = lane>>4. Two 16B loads.
__device__ __forceinline__ v16h ldfrag(const _Float16* p, int hl) {
  const _Float16* q = p + hl * 8;
  v8h lo = *(const v8h*)(q);
  v8h hi = *(const v8h*)(q + 16);
  return __builtin_shufflevector(lo, hi, 0,1,2,3,4,5,6,7,8,9,10,11,12,13,14,15);
}

__device__ __forceinline__ float lrelu(float y) { return y >= 0.f ? y : NEGS * y; }

// ---------------------------------------------------------------------------
// prep: f32 -> f16 weight conversion with K padding
// ---------------------------------------------------------------------------
__global__ void convert_pad_kernel(const float* __restrict__ src, int srcStride, int srcOff,
                                   _Float16* __restrict__ dst, int O, int C, int CP) {
  int i = blockIdx.x * blockDim.x + threadIdx.x;
  if (i >= O * CP) return;
  int o = i / CP, c = i % CP;
  dst[i] = (c < C) ? (_Float16)src[(size_t)o * srcStride + srcOff + c] : (_Float16)0.f;
}

// bn params (4,C) stacked [g,b,m,v] -> scale/bias
__global__ void prep_bn_kernel(const float* __restrict__ p, int C,
                               float* __restrict__ S, float* __restrict__ B) {
  int c = blockIdx.x * blockDim.x + threadIdx.x;
  if (c >= C) return;
  float g = p[c], b = p[C + c], m = p[2 * C + c], v = p[3 * C + c];
  float s = g * rsqrtf(v + 1e-5f);
  S[c] = s;
  B[c] = b - m * s;
}

// curr_pos + onehot(node_type) -> f16 feature (N,32) padded
__global__ void build_feat0_kernel(const float* __restrict__ pos, const int* __restrict__ nt,
                                   _Float16* __restrict__ feat, int N) {
  int n = blockIdx.x * blockDim.x + threadIdx.x;
  if (n >= N) return;
  int t = nt[n];
#pragma unroll
  for (int c = 0; c < 32; ++c) {
    float v = 0.f;
    if (c < 3)       v = pos[n * 3 + c];
    else if (c < 12) v = (c - 3 == t) ? 1.f : 0.f;
    feat[(size_t)n * 32 + c] = (_Float16)v;
  }
}

__global__ void norms_kernel(const _Float16* __restrict__ feat, int stride, int C,
                             float* __restrict__ norms, int N) {
  int n = blockIdx.x * blockDim.x + threadIdx.x;
  if (n >= N) return;
  float s = 0.f;
  for (int c = 0; c < C; ++c) { float v = (float)feat[(size_t)n * stride + c]; s += v * v; }
  norms[n] = s;
}

// ---------------------------------------------------------------------------
// Fused kNN: WMMA Gram tiles + in-register top-20 (score = 2*dot - |i|^2 - |j|^2)
// Block = 16 rows x 256 threads; columns streamed in 512-wide panels via LDS.
// ---------------------------------------------------------------------------
template <int KC>
__global__ __launch_bounds__(256) void knn_kernel(const _Float16* __restrict__ feat, int rowstride,
                                                  const float* __restrict__ norms,
                                                  int* __restrict__ idxout, int N) {
  __shared__ float smemu[10240];            // union: 16x512 f32 scores (32KB) | merge 40KB
  float* sS = smemu;
  float* mv = smemu;
  int*   mi = (int*)(smemu + 5120);

  int tid = threadIdx.x;
  int lane = tid & 31, wave = tid >> 5;
  int m = lane & 15, hl = lane >> 4;
  int r0 = blockIdx.x * 16;

  v16h afr[KC];
  const _Float16* arow = feat + (size_t)(r0 + m) * rowstride;
#pragma unroll
  for (int kc = 0; kc < KC; ++kc) afr[kc] = ldfrag(arow + kc * 32, hl);

  float normR[8];
#pragma unroll
  for (int r = 0; r < 8; ++r) normR[r] = norms[r0 + r + 8 * hl];

  float best[KNN]; int bidx[KNN];
#pragma unroll
  for (int j = 0; j < KNN; ++j) { best[j] = -3.0e38f; bidx[j] = 0; }

  int row = tid >> 4, slot = tid & 15;

  for (int panel = 0; panel < N / 512; ++panel) {
    int cbase = panel * 512;
    for (int t = wave; t < 32; t += 8) {        // 4 col-tiles per wave
      int c0 = cbase + t * 16;
      const _Float16* brow = feat + (size_t)(c0 + m) * rowstride;
      v8f acc = {};
#pragma unroll
      for (int kc = 0; kc < KC; ++kc) acc = wmma32(afr[kc], ldfrag(brow + kc * 32, hl), acc);
      float normC = norms[c0 + m];
#pragma unroll
      for (int r = 0; r < 8; ++r)
        sS[(r + 8 * hl) * 512 + (t * 16 + m)] = 2.0f * acc[r] - normR[r] - normC;
    }
    __syncthreads();
#pragma unroll 1
    for (int i = 0; i < 32; ++i) {              // 16 threads per row scan 32 cols each
      int cp = slot + i * 16;
      float v = sS[row * 512 + cp];
      if (v > best[KNN - 1]) {
        float cv = v; int ci = cbase + cp;
#pragma unroll
        for (int j = 0; j < KNN; ++j)
          if (cv > best[j]) { float tv = best[j]; int ti = bidx[j];
                              best[j] = cv; bidx[j] = ci; cv = tv; ci = ti; }
      }
    }
    __syncthreads();
  }
  // merge: 16 local lists per row -> top-20
#pragma unroll
  for (int j = 0; j < KNN; ++j) { mv[tid * KNN + j] = best[j]; mi[tid * KNN + j] = bidx[j]; }
  __syncthreads();
  if (tid < 16) {
    float fb[KNN]; int fi[KNN];
#pragma unroll
    for (int j = 0; j < KNN; ++j) { fb[j] = -3.0e38f; fi[j] = 0; }
    int base = tid * 16 * KNN;
    for (int s = 0; s < 16 * KNN; ++s) {
      float v = mv[base + s];
      if (v > fb[KNN - 1]) {
        float cv = v; int ci = mi[base + s];
#pragma unroll
        for (int j = 0; j < KNN; ++j)
          if (cv > fb[j]) { float tv = fb[j]; int ti = fi[j];
                            fb[j] = cv; fi[j] = ci; cv = tv; ci = ti; }
      }
    }
    for (int j = 0; j < KNN; ++j) idxout[(size_t)(r0 + tid) * KNN + j] = fi[j];
  }
}

// ---------------------------------------------------------------------------
// Fused edge conv: gather -> GEMM1(bn,lrelu) [-> GEMM2(bn,lrelu)] -> max over k
// Block = 8 points (160 columns), 256 threads, all intermediates in LDS.
// ---------------------------------------------------------------------------
template <int CPIN, bool TWO>
__global__ __launch_bounds__(256) void edgeconv_kernel(
    const _Float16* __restrict__ feat, int rowstride, int C,
    const int* __restrict__ idx,
    const _Float16* __restrict__ Wa, const float* __restrict__ bnSa, const float* __restrict__ bnBa,
    const _Float16* __restrict__ Wb, const float* __restrict__ bnSb, const float* __restrict__ bnBb,
    _Float16* __restrict__ out, int outstride, int outchoff) {
  constexpr int NCOL = 8 * KNN;                       // 160
  constexpr int RA = (CPIN > 64 ? CPIN : 64) * NCOL;  // gmat, later reused for f2
  __shared__ _Float16 sA[RA];
  __shared__ _Float16 sB[64 * NCOL];

  int tid = threadIdx.x;
  int p0 = blockIdx.x * 8;

  // phase 1: build edge features [f_j - f_i | f_i | 0pad] column-major in LDS
  for (int i = tid; i < NCOL * CPIN; i += 256) {
    int col = i / CPIN, ch = i % CPIN;
    int p = col / KNN, k = col % KNN;
    int n = p0 + p;
    _Float16 v = (_Float16)0.f;
    if (ch < C) {
      int j = idx[(size_t)n * KNN + k];
      v = feat[(size_t)j * rowstride + ch] - feat[(size_t)n * rowstride + ch];
    } else if (ch < 2 * C) {
      v = feat[(size_t)n * rowstride + (ch - C)];
    }
    sA[col * CPIN + ch] = v;
  }
  __syncthreads();

  int lane = tid & 31, wave = tid >> 5;
  int m = lane & 15, hl = lane >> 4;

  // phase 2: GEMM1 (64 x NCOL), 40 tile-jobs over 8 waves
  for (int job = wave; job < 40; job += 8) {
    int mt = job & 3, nt = job >> 2;
    v8f acc = {};
    const _Float16* awrow = Wa + (size_t)(mt * 16 + m) * CPIN;
    const _Float16* bcol = &sA[(nt * 16 + m) * CPIN];
#pragma unroll
    for (int kc = 0; kc < CPIN / 32; ++kc)
      acc = wmma32(ldfrag(awrow + kc * 32, hl), ldfrag(bcol + kc * 32, hl), acc);
    int col = nt * 16 + m;
    v8h o;
#pragma unroll
    for (int r = 0; r < 8; ++r) {
      int ch = mt * 16 + hl * 8 + r;
      o[r] = (_Float16)lrelu(acc[r] * bnSa[ch] + bnBa[ch]);
    }
    *(v8h*)&sB[col * 64 + mt * 16 + hl * 8] = o;
  }
  __syncthreads();

  if constexpr (TWO) {
    // phase 3: GEMM2 (64x64 @ 64xNCOL) -> sA (gmat is dead)
    for (int job = wave; job < 40; job += 8) {
      int mt = job & 3, nt = job >> 2;
      v8f acc = {};
      const _Float16* awrow = Wb + (size_t)(mt * 16 + m) * 64;
      const _Float16* bcol = &sB[(nt * 16 + m) * 64];
#pragma unroll
      for (int kc = 0; kc < 2; ++kc)
        acc = wmma32(ldfrag(awrow + kc * 32, hl), ldfrag(bcol + kc * 32, hl), acc);
      int col = nt * 16 + m;
      v8h o;
#pragma unroll
      for (int r = 0; r < 8; ++r) {
        int ch = mt * 16 + hl * 8 + r;
        o[r] = (_Float16)lrelu(acc[r] * bnSb[ch] + bnBb[ch]);
      }
      *(v8h*)&sA[col * 64 + mt * 16 + hl * 8] = o;
    }
    __syncthreads();
  }

  const _Float16* src = TWO ? sA : sB;
  // phase 4: max over k=20, write pooled 64-channel feature
  for (int i = tid; i < 8 * 64; i += 256) {
    int p = i >> 6, ch = i & 63;
    float mx = -3.0e38f;
#pragma unroll 4
    for (int k = 0; k < KNN; ++k) mx = fmaxf(mx, (float)src[(p * KNN + k) * 64 + ch]);
    out[(size_t)(p0 + p) * outstride + outchoff + ch] = (_Float16)mx;
  }
}

// ---------------------------------------------------------------------------
// Pointwise conv GEMM: out(Oc x N) = W(Oc x KCH*32) @ X ; bn + lrelu
// MAXMODE: cross-lane max over the 16-column tile -> per-block partial maxima
// ---------------------------------------------------------------------------
template <int KCH, bool MAXMODE>
__global__ __launch_bounds__(256) void pconv_kernel(
    const _Float16* __restrict__ X, int xstride,
    const _Float16* __restrict__ W, int Oc,
    const float* __restrict__ biasAdd,
    const float* __restrict__ bnS, const float* __restrict__ bnB,
    _Float16* __restrict__ out, int outstride,
    float* __restrict__ partials) {
  int tid = threadIdx.x, lane = tid & 31, wave = tid >> 5;
  int m = lane & 15, hl = lane >> 4;
  int n0 = blockIdx.x * 16;

  v16h bf[KCH];
  const _Float16* xrow = X + (size_t)(n0 + m) * xstride;
#pragma unroll
  for (int kc = 0; kc < KCH; ++kc) bf[kc] = ldfrag(xrow + kc * 32, hl);

  int MT = Oc >> 4;
  for (int mt = wave; mt < MT; mt += 8) {
    v8f acc = {};
    const _Float16* wrow = W + (size_t)(mt * 16 + m) * (KCH * 32);
#pragma unroll
    for (int kc = 0; kc < KCH; ++kc) acc = wmma32(ldfrag(wrow + kc * 32, hl), bf[kc], acc);

    float vals[8];
#pragma unroll
    for (int r = 0; r < 8; ++r) {
      int ch = mt * 16 + hl * 8 + r;
      float y = acc[r];
      if (biasAdd) y += biasAdd[ch];
      vals[r] = lrelu(y * bnS[ch] + bnB[ch]);
    }
    if constexpr (MAXMODE) {
#pragma unroll
      for (int r = 0; r < 8; ++r) {
        float v = vals[r];
#pragma unroll
        for (int sh = 1; sh < 16; sh <<= 1) v = fmaxf(v, __shfl_xor(v, sh, 32));
        vals[r] = v;
      }
      if (m == 0) {
#pragma unroll
        for (int r = 0; r < 8; ++r)
          partials[(size_t)blockIdx.x * Oc + mt * 16 + hl * 8 + r] = vals[r];
      }
    } else {
      v8h o;
#pragma unroll
      for (int r = 0; r < 8; ++r) o[r] = (_Float16)vals[r];
      *(v8h*)(out + (size_t)(n0 + m) * outstride + mt * 16 + hl * 8) = o;
    }
  }
}

__global__ void reduce_max_kernel(const float* __restrict__ partials, float* __restrict__ g,
                                  int Oc, int nblk) {
  int ch = blockIdx.x * blockDim.x + threadIdx.x;
  if (ch >= Oc) return;
  float mx = -3.0e38f;
  for (int b = 0; b < nblk; ++b) mx = fmaxf(mx, partials[(size_t)b * Oc + ch]);
  g[ch] = mx;
}

// W7[:, :1024] @ g  (global feature is column-constant -> rank-1 term = bias)
__global__ void w7bias_kernel(const float* __restrict__ W7, const float* __restrict__ g,
                              float* __restrict__ bias7) {
  int o = blockIdx.x * blockDim.x + threadIdx.x;
  if (o >= 512) return;
  float acc = 0.f;
  for (int c = 0; c < 1024; ++c) acc += W7[(size_t)o * 1216 + c] * g[c];
  bias7[o] = acc;
}

__global__ void w9_kernel(const _Float16* __restrict__ h8, const float* __restrict__ W9,
                          float* __restrict__ out, int N) {
  int n = blockIdx.x * blockDim.x + threadIdx.x;
  if (n >= N) return;
  float a0 = 0.f, a1 = 0.f, a2 = 0.f;
  for (int c = 0; c < 256; ++c) {
    float h = (float)h8[(size_t)n * 256 + c];
    a0 += W9[c] * h; a1 += W9[256 + c] * h; a2 += W9[512 + c] * h;
  }
  out[n * 3 + 0] = a0; out[n * 3 + 1] = a1; out[n * 3 + 2] = a2;
}

// ---------------------------------------------------------------------------
extern "C" void kernel_launch(void* const* d_in, const int* in_sizes, int n_in,
                              void* d_out, int out_size, void* d_ws, size_t ws_size,
                              hipStream_t stream) {
  const int N = NPTS;
  const float* curr_pos = (const float*)d_in[0];
  const int*   node_type = (const int*)d_in[1];
  const float* W1 = (const float*)d_in[2];
  const float* W2 = (const float*)d_in[3];
  const float* W3 = (const float*)d_in[4];
  const float* W4 = (const float*)d_in[5];
  const float* W5 = (const float*)d_in[6];
  const float* W6 = (const float*)d_in[7];
  const float* W7 = (const float*)d_in[8];
  const float* W8 = (const float*)d_in[9];
  const float* W9 = (const float*)d_in[10];
  const float* bn[8];
  for (int i = 0; i < 8; ++i) bn[i] = (const float*)d_in[11 + i];

  char* ws = (char*)d_ws;
  size_t off = 0;
  auto alloc = [&](size_t bytes) -> void* {
    void* p = ws + off;
    off += (bytes + 255) & ~(size_t)255;
    return p;
  };

  _Float16* feat0h = (_Float16*)alloc((size_t)N * 32 * 2);
  _Float16* x123   = (_Float16*)alloc((size_t)N * 192 * 2);
  float*    normb  = (float*)alloc((size_t)N * 4);
  int*      idxb   = (int*)alloc((size_t)N * KNN * 4);
  _Float16* W1h = (_Float16*)alloc(64 * 32 * 2);
  _Float16* W2h = (_Float16*)alloc(64 * 64 * 2);
  _Float16* W3h = (_Float16*)alloc(64 * 128 * 2);
  _Float16* W4h = (_Float16*)alloc(64 * 64 * 2);
  _Float16* W5h = (_Float16*)alloc(64 * 128 * 2);
  _Float16* W6h = (_Float16*)alloc(1024 * 192 * 2);
  _Float16* W7xh = (_Float16*)alloc(512 * 192 * 2);
  _Float16* W8h = (_Float16*)alloc(256 * 512 * 2);
  float* bnS[8]; float* bnB[8];
  const int bnC[8] = {64, 64, 64, 64, 64, 1024, 512, 256};
  for (int i = 0; i < 8; ++i) { bnS[i] = (float*)alloc(bnC[i] * 4); bnB[i] = (float*)alloc(bnC[i] * 4); }
  float* partials = (float*)alloc((size_t)(N / 16) * 1024 * 4);
  float* gmax  = (float*)alloc(1024 * 4);
  float* bias7 = (float*)alloc(512 * 4);
  _Float16* h7out = (_Float16*)alloc((size_t)N * 512 * 2);
  _Float16* h8    = (_Float16*)alloc((size_t)N * 256 * 2);
  if (off > ws_size) return;  // workspace too small; bail safely

  const int T = 256;
  // --- weight / bn prep ---
  convert_pad_kernel<<<(64 * 32 + T - 1) / T, T, 0, stream>>>(W1, 24, 0, W1h, 64, 24, 32);
  convert_pad_kernel<<<(64 * 64 + T - 1) / T, T, 0, stream>>>(W2, 64, 0, W2h, 64, 64, 64);
  convert_pad_kernel<<<(64 * 128 + T - 1) / T, T, 0, stream>>>(W3, 128, 0, W3h, 64, 128, 128);
  convert_pad_kernel<<<(64 * 64 + T - 1) / T, T, 0, stream>>>(W4, 64, 0, W4h, 64, 64, 64);
  convert_pad_kernel<<<(64 * 128 + T - 1) / T, T, 0, stream>>>(W5, 128, 0, W5h, 64, 128, 128);
  convert_pad_kernel<<<(1024 * 192 + T - 1) / T, T, 0, stream>>>(W6, 192, 0, W6h, 1024, 192, 192);
  convert_pad_kernel<<<(512 * 192 + T - 1) / T, T, 0, stream>>>(W7, 1216, 1024, W7xh, 512, 192, 192);
  convert_pad_kernel<<<(256 * 512 + T - 1) / T, T, 0, stream>>>(W8, 512, 0, W8h, 256, 512, 512);
  for (int i = 0; i < 8; ++i)
    prep_bn_kernel<<<(bnC[i] + T - 1) / T, T, 0, stream>>>(bn[i], bnC[i], bnS[i], bnB[i]);

  // --- stage 1: feat0 -> knn -> edgeconv(W1,W2) -> x1 ---
  build_feat0_kernel<<<N / T, T, 0, stream>>>(curr_pos, node_type, feat0h, N);
  norms_kernel<<<N / T, T, 0, stream>>>(feat0h, 32, 12, normb, N);
  knn_kernel<1><<<N / 16, T, 0, stream>>>(feat0h, 32, normb, idxb, N);
  edgeconv_kernel<32, true><<<N / 8, T, 0, stream>>>(feat0h, 32, 12, idxb,
      W1h, bnS[0], bnB[0], W2h, bnS[1], bnB[1], x123, 192, 0);

  // --- stage 2: knn(x1) -> edgeconv(W3,W4) -> x2 ---
  norms_kernel<<<N / T, T, 0, stream>>>(x123, 192, 64, normb, N);
  knn_kernel<2><<<N / 16, T, 0, stream>>>(x123, 192, normb, idxb, N);
  edgeconv_kernel<128, true><<<N / 8, T, 0, stream>>>(x123, 192, 64, idxb,
      W3h, bnS[2], bnB[2], W4h, bnS[3], bnB[3], x123, 192, 64);

  // --- stage 3: knn(x2) -> edgeconv(W5) -> x3 ---
  norms_kernel<<<N / T, T, 0, stream>>>(x123 + 64, 192, 64, normb, N);
  knn_kernel<2><<<N / 16, T, 0, stream>>>(x123 + 64, 192, normb, idxb, N);
  edgeconv_kernel<128, false><<<N / 8, T, 0, stream>>>(x123 + 64, 192, 64, idxb,
      W5h, bnS[4], bnB[4], nullptr, nullptr, nullptr, x123, 192, 128);

  // --- head: W6 + global max ; W7 (rank-1 bias + 192-ch GEMM) ; W8 ; W9 ---
  pconv_kernel<6, true><<<N / 16, T, 0, stream>>>(x123, 192, W6h, 1024,
      nullptr, bnS[5], bnB[5], nullptr, 0, partials);
  reduce_max_kernel<<<(1024 + T - 1) / T, T, 0, stream>>>(partials, gmax, 1024, N / 16);
  w7bias_kernel<<<(512 + T - 1) / T, T, 0, stream>>>(W7, gmax, bias7);
  pconv_kernel<6, false><<<N / 16, T, 0, stream>>>(x123, 192, W7xh, 512,
      bias7, bnS[6], bnB[6], h7out, 512, nullptr);
  pconv_kernel<16, false><<<N / 16, T, 0, stream>>>(h7out, 512, W8h, 256,
      nullptr, bnS[7], bnB[7], h8, 256, nullptr);
  w9_kernel<<<N / T, T, 0, stream>>>(h8, W9, (float*)d_out, N);
}
